// VecIntInitializer_20555713479010
// MI455X (gfx1250) — matrix operand order repaired and checked by reading, not compiled, not run on hardware
//
// MI455X (gfx1250, wave32) forward pass for the trajectory-prediction model.
// All dense math runs on v_wmma_f32_16x16x32_f16 (f16 in, f32 accumulate).
// Weights are pre-transposed+converted to fp16 once per launch so GEMM
// B-fragments are contiguous 16-byte loads (no LDS staging, no barriers,
// 2x-unrolled ping-pong software-pipelined hot loop). Attention is
// flash-style (never materializes the 4096x4096 score matrix), split over
// 4 key partitions, merged by a combiner; V-fragment loads are hoisted
// ahead of the softmax.
//
// Parameter pointer order assumption (jax tree flatten: top-level dict in
// insertion order, nested dicts sorted by key, lists in order, 'W' < 'b'):
//   0:x 1:intention 2:past_similarity 3:mask, then params sorted:
//   ego_int(4) ego_mean(4) ego_scale(4) ego_sim(4) ego_var(4)
//   int_dec(6) int_social(34) mean_dec(6) scale_dec(6) scale_enc(6)
//   sim_dec(6) sim_social(34) social(34) var_dec(6)   => n_in = 162
// tlayer leaf order (sorted): f1.W f1.b f2.W f2.b k.W k.b ln1_b ln1_g
//   ln2_b ln2_g o.W o.b q.W q.b v.W v.b

#include <hip/hip_runtime.h>
#include <math.h>

#define BB      4096   // batch / number of agents (attention sequence length)
#define NHEADS  2
#define HDIM    128
#define DM      256
#define KPART   4      // key partitions for flash attention split-K

typedef __attribute__((ext_vector_type(16))) _Float16 v16h;
typedef __attribute__((ext_vector_type(8)))  _Float16 v8h;
typedef __attribute__((ext_vector_type(8)))  float    v8f;

static __device__ __forceinline__ v16h cat8(v8h lo, v8h hi) {
  return __builtin_shufflevector(lo, hi, 0,1,2,3,4,5,6,7,8,9,10,11,12,13,14,15);
}

static __device__ __forceinline__ v16h cvt16(float4 f0, float4 f1, float4 f2, float4 f3) {
  v16h a;
  a[0]  = (_Float16)f0.x; a[1]  = (_Float16)f0.y; a[2]  = (_Float16)f0.z; a[3]  = (_Float16)f0.w;
  a[4]  = (_Float16)f1.x; a[5]  = (_Float16)f1.y; a[6]  = (_Float16)f1.z; a[7]  = (_Float16)f1.w;
  a[8]  = (_Float16)f2.x; a[9]  = (_Float16)f2.y; a[10] = (_Float16)f2.z; a[11] = (_Float16)f2.w;
  a[12] = (_Float16)f3.x; a[13] = (_Float16)f3.y; a[14] = (_Float16)f3.z; a[15] = (_Float16)f3.w;
  return a;
}

// A-fragment: 4x float4 + packed f16 convert (two contiguous 8-K runs)
static __device__ __forceinline__ v16h load_a(const float* p) {
  return cvt16(*(const float4*)(p),      *(const float4*)(p + 4),
               *(const float4*)(p + 16), *(const float4*)(p + 20));
}
// B-fragment: two contiguous b128 fp16 loads
static __device__ __forceinline__ v16h load_b(const _Float16* w) {
  return cat8(*(const v8h*)w, *(const v8h*)(w + 16));
}

// Weight transpose + fp32->fp16 convert: W (KxM row-major) -> WT (MxK row-major)
__global__ void wcvt_kernel(const float* __restrict__ W, _Float16* __restrict__ WT,
                            int K, int M)
{
  int i = blockIdx.x * blockDim.x + threadIdx.x;
  if (i >= K * M) return;
  int k = i / M, m = i % M;
  WT[(size_t)m * K + k] = (_Float16)W[i];
}

// ---------------------------------------------------------------------------
// GEMM: Y = act(X @ W + b).  X: NxK fp32, WT: MxK fp16 (pre-transposed),
// Y: NxM fp32.  One wave computes a 16x64 tile (4 accumulators, A reused 4x).
// 2x-unrolled ping-pong pipeline: while the WMMAs of buffer 0 execute,
// buffer 1's 12 vector loads are in flight (and vice versa) - no register
// rotation copies, only partial load waits.
// A-frag: lane L elem e -> X[nbase+(L&15)][kb + (e&7)+16*(e>>3)+8*(L>>4)]
// B-frag: lane L elem e -> WT[col=mbase+16m+(L&15)][same-K]
// C/D   : lane L reg  r -> Y[nbase + r + 8*(L>>4)][col]
// N must be a multiple of 16 (true for all launches here).
// ---------------------------------------------------------------------------
__global__ __launch_bounds__(32) void gemm_wmma_kernel(
    const float* __restrict__ X, const _Float16* __restrict__ WT,
    const float* __restrict__ bias, float* __restrict__ Y,
    int N, int K, int M, int act)
{
  const int lane  = threadIdx.x;
  const int mbase = blockIdx.x * 64;
  const int nbase = blockIdx.y * 16;
  const int g     = lane >> 4;
  const int ln    = lane & 15;
  const int koff  = g * 8;

  const float* xrow = X + (size_t)(nbase + ln) * K + koff;
  const _Float16* wp[4];
#pragma unroll
  for (int m = 0; m < 4; ++m) {
    int col = mbase + m * 16 + ln;
    int colc = (col < M) ? col : (M - 1);     // clamp: always-valid loads
    wp[m] = WT + (size_t)colc * K + koff;
  }

  v8f acc[4];
#pragma unroll
  for (int m = 0; m < 4; ++m) acc[m] = (v8f){};

  const int Kfull = K & ~31;
  if (Kfull > 0) {
    // ping-pong software pipeline, unrolled by 2 (no rotation copies)
    v16h a0 = load_a(xrow);
    v16h b0[4];
#pragma unroll
    for (int m = 0; m < 4; ++m) b0[m] = load_b(wp[m]);
    int kb = 32;
    for (; kb + 32 < Kfull; kb += 64) {
      v16h a1 = load_a(xrow + kb);
      v16h b1[4];
#pragma unroll
      for (int m = 0; m < 4; ++m) b1[m] = load_b(wp[m] + kb);
#pragma unroll
      for (int m = 0; m < 4; ++m)
        acc[m] = __builtin_amdgcn_wmma_f32_16x16x32_f16(false, a0, false, b0[m],
                                                        (short)0, acc[m], false, false);
      a0 = load_a(xrow + kb + 32);
#pragma unroll
      for (int m = 0; m < 4; ++m) b0[m] = load_b(wp[m] + kb + 32);
#pragma unroll
      for (int m = 0; m < 4; ++m)
        acc[m] = __builtin_amdgcn_wmma_f32_16x16x32_f16(false, a1, false, b1[m],
                                                        (short)0, acc[m], false, false);
    }
    if (kb < Kfull) {            // one more pair: kb == Kfull - 32
      v16h a1 = load_a(xrow + kb);
      v16h b1[4];
#pragma unroll
      for (int m = 0; m < 4; ++m) b1[m] = load_b(wp[m] + kb);
#pragma unroll
      for (int m = 0; m < 4; ++m)
        acc[m] = __builtin_amdgcn_wmma_f32_16x16x32_f16(false, a0, false, b0[m],
                                                        (short)0, acc[m], false, false);
#pragma unroll
      for (int m = 0; m < 4; ++m)
        acc[m] = __builtin_amdgcn_wmma_f32_16x16x32_f16(false, a1, false, b1[m],
                                                        (short)0, acc[m], false, false);
    } else {                     // single pending buffer
#pragma unroll
      for (int m = 0; m < 4; ++m)
        acc[m] = __builtin_amdgcn_wmma_f32_16x16x32_f16(false, a0, false, b0[m],
                                                        (short)0, acc[m], false, false);
    }
  }
  if (Kfull < K) {                             // K tail (scalar, once per tile)
    const float* xr = X + (size_t)(nbase + ln) * K;   // unshifted row
    v16h a = {};
#pragma unroll
    for (int e = 0; e < 16; ++e) {
      int kg = Kfull + koff + ((e < 8) ? e : (8 + e));   // e>=8 -> koff+16+(e-8)
      a[e] = (kg < K) ? (_Float16)xr[kg] : (_Float16)0.f;
    }
#pragma unroll
    for (int m = 0; m < 4; ++m) {
      const _Float16* wr = wp[m] - koff;
      v16h b = {};
#pragma unroll
      for (int e = 0; e < 16; ++e) {
        int kg = Kfull + koff + ((e < 8) ? e : (8 + e));
        b[e] = (kg < K) ? wr[kg] : (_Float16)0.f;
      }
      acc[m] = __builtin_amdgcn_wmma_f32_16x16x32_f16(false, a, false, b,
                                                      (short)0, acc[m], false, false);
    }
  }
#pragma unroll
  for (int m = 0; m < 4; ++m) {
    int col = mbase + m * 16 + ln;
    if (col < M) {
      float bv = bias ? bias[col] : 0.f;
#pragma unroll
      for (int r = 0; r < 8; ++r) {
        int row = nbase + r + 8 * g;
        float v = acc[m][r] + bv;
        if (act == 1) v = fmaxf(v, 0.f);
        Y[(size_t)row * M + col] = v;
      }
    }
  }
}

// ---------------------------------------------------------------------------
// Pack q,k to fp16 row-major; v to fp16 transposed (DM x BB) so flash-attn
// B-fragments are contiguous 16-byte loads.
// ---------------------------------------------------------------------------
__global__ void pack_qkv_kernel(const float* __restrict__ q, const float* __restrict__ k,
                                const float* __restrict__ v, _Float16* __restrict__ qh,
                                _Float16* __restrict__ kh, _Float16* __restrict__ vt)
{
  int i = blockIdx.x * blockDim.x + threadIdx.x;
  if (i >= BB * DM) return;
  int r = i >> 8, c = i & 255;
  qh[i] = (_Float16)q[i];
  kh[i] = (_Float16)k[i];
  vt[(size_t)c * BB + r] = (_Float16)v[i];
}

// ---------------------------------------------------------------------------
// Flash attention partial: one wave per (16-query tile, head, key partition).
// Produces unnormalized O accumulator plus per-row (m, l) for the combiner.
// V-fragment loads are issued before the softmax chain so the P.V WMMAs
// never wait on memory.
// ---------------------------------------------------------------------------
__global__ __launch_bounds__(32) void flash_attn_kernel(
    const _Float16* __restrict__ Qh, const _Float16* __restrict__ Kh,
    const _Float16* __restrict__ VT, const float* __restrict__ mask,
    float* __restrict__ Apart, float* __restrict__ Mbuf, float* __restrict__ Lbuf)
{
  __shared__ _Float16 Ps[16 * 32];
  const int lane  = threadIdx.x;
  const int qbase = blockIdx.x * 16;
  const int head  = blockIdx.y;
  const int part  = blockIdx.z;
  const int g     = lane >> 4;
  const int ln    = lane & 15;
  const int koff  = g * 8;
  const float scale = 0.088388347648318447f;   // 1/sqrt(128)

  v16h qa[4];
  {
    const _Float16* qrow = Qh + (size_t)(qbase + ln) * DM + head * HDIM + koff;
#pragma unroll
    for (int t = 0; t < 4; ++t)
      qa[t] = load_b(qrow + t * 32);   // same two-run pattern as B-frag
  }
  v8f zero = {};
  v8f o[8];
#pragma unroll
  for (int t = 0; t < 8; ++t) o[t] = zero;
  float mrow[8], lrow[8];
#pragma unroll
  for (int r = 0; r < 8; ++r) { mrow[r] = -3.0e38f; lrow[r] = 0.f; }

  const int k0 = part * (BB / KPART);
  const int k1 = k0 + (BB / KPART);
  for (int j = k0; j < k1; j += 32) {
    // ---- S = Q.K^T for two 16-key sub-tiles ----
    v8f S[2];
#pragma unroll
    for (int sub = 0; sub < 2; ++sub) {
      int key = j + sub * 16 + ln;
      const _Float16* krow = Kh + (size_t)key * DM + head * HDIM + koff;
      v8f s = zero;
#pragma unroll
      for (int t = 0; t < 4; ++t) {
        v16h kb = load_b(krow + t * 32);
        s = __builtin_amdgcn_wmma_f32_16x16x32_f16(false, qa[t], false, kb,
                                                   (short)0, s, false, false);
      }
      S[sub] = s;
    }
    // ---- issue V fragment loads early (independent of softmax) ----
    v16h vf[8];
#pragma unroll
    for (int t = 0; t < 8; ++t)
      vf[t] = load_b(VT + (size_t)(head * HDIM + t * 16 + ln) * BB + j + koff);
    // ---- mask + online softmax over this 32-key block ----
#pragma unroll
    for (int sub = 0; sub < 2; ++sub) {
#pragma unroll
      for (int r = 0; r < 8; ++r) {
        int qi = qbase + r + 8 * g;
        int key = j + sub * 16 + ln;
        float mv = mask[(size_t)qi * BB + key];
        float madd = (mv == 0.f) ? -3.0e38f : ((mv == 1.f) ? 0.f : mv);
        S[sub][r] = S[sub][r] * scale + madd;
      }
    }
    float pa[8], pb[8], al[8];
#pragma unroll
    for (int r = 0; r < 8; ++r) {
      float mx = fmaxf(S[0][r], S[1][r]);
#pragma unroll
      for (int od = 1; od < 16; od <<= 1) mx = fmaxf(mx, __shfl_xor(mx, od, 32));
      float mnew = fmaxf(mrow[r], mx);
      al[r] = __expf(mrow[r] - mnew);
      pa[r] = __expf(S[0][r] - mnew);
      pb[r] = __expf(S[1][r] - mnew);
      float rs = pa[r] + pb[r];
#pragma unroll
      for (int od = 1; od < 16; od <<= 1) rs += __shfl_xor(rs, od, 32);
      lrow[r] = lrow[r] * al[r] + rs;
      mrow[r] = mnew;
    }
#pragma unroll
    for (int t = 0; t < 8; ++t)
#pragma unroll
      for (int r = 0; r < 8; ++r) o[t][r] *= al[r];
    // ---- C-layout P -> row-major 16x32 fp16 in LDS -> A-fragment ----
    __syncthreads();
#pragma unroll
    for (int r = 0; r < 8; ++r) {
      Ps[(r + 8 * g) * 32 + ln]      = (_Float16)pa[r];
      Ps[(r + 8 * g) * 32 + 16 + ln] = (_Float16)pb[r];
    }
    __syncthreads();
    v16h pf = cat8(*(const v8h*)&Ps[ln * 32 + koff],
                   *(const v8h*)&Ps[ln * 32 + koff + 16]);
#pragma unroll
    for (int t = 0; t < 8; ++t)
      o[t] = __builtin_amdgcn_wmma_f32_16x16x32_f16(false, pf, false, vf[t],
                                                    (short)0, o[t], false, false);
  }
#pragma unroll
  for (int t = 0; t < 8; ++t)
#pragma unroll
    for (int r = 0; r < 8; ++r) {
      int row = qbase + r + 8 * g;
      int col = head * HDIM + t * 16 + ln;
      Apart[((size_t)part * BB + row) * DM + col] = o[t][r];
    }
  if (ln == 0) {
#pragma unroll
    for (int r = 0; r < 8; ++r) {
      int row = qbase + r + 8 * g;
      size_t idx = ((size_t)part * NHEADS + head) * BB + row;
      Mbuf[idx] = mrow[r];
      Lbuf[idx] = lrow[r];
    }
  }
}

__global__ void flash_combine_kernel(const float* __restrict__ Apart,
                                     const float* __restrict__ Mbuf,
                                     const float* __restrict__ Lbuf,
                                     float* __restrict__ O)
{
  int i = blockIdx.x * blockDim.x + threadIdx.x;
  if (i >= BB * DM) return;
  int row = i >> 8, col = i & 255, head = col >> 7;
  float mp[KPART], lp[KPART], m = -3.0e38f;
#pragma unroll
  for (int p = 0; p < KPART; ++p) {
    size_t idx = ((size_t)p * NHEADS + head) * BB + row;
    mp[p] = Mbuf[idx]; lp[p] = Lbuf[idx];
    m = fmaxf(m, mp[p]);
  }
  float denom = 0.f, acc = 0.f;
#pragma unroll
  for (int p = 0; p < KPART; ++p) {
    float w = __expf(mp[p] - m);
    denom += lp[p] * w;
    acc += Apart[(size_t)p * BB * DM + i] * w;
  }
  O[i] = acc / denom;
}

// h = LayerNorm(h + resid) * gamma + beta  (D = 256, one wave per row)
__global__ __launch_bounds__(256) void ln_res_kernel(
    float* __restrict__ h, const float* __restrict__ resid,
    const float* __restrict__ gamma, const float* __restrict__ beta, int rows)
{
  int w = threadIdx.x >> 5, lane = threadIdx.x & 31;
  int row = blockIdx.x * 8 + w;
  if (row >= rows) return;
  float v[8]; float s = 0.f;
#pragma unroll
  for (int kk = 0; kk < 8; ++kk) {
    int c = lane + 32 * kk;
    v[kk] = h[(size_t)row * DM + c] + resid[(size_t)row * DM + c];
    s += v[kk];
  }
#pragma unroll
  for (int od = 16; od > 0; od >>= 1) s += __shfl_xor(s, od, 32);
  float mean = s * (1.f / 256.f);
  float vs = 0.f;
#pragma unroll
  for (int kk = 0; kk < 8; ++kk) { float d = v[kk] - mean; vs += d * d; }
#pragma unroll
  for (int od = 16; od > 0; od >>= 1) vs += __shfl_xor(vs, od, 32);
  float rstd = rsqrtf(vs * (1.f / 256.f) + 1e-5f);
#pragma unroll
  for (int kk = 0; kk < 8; ++kk) {
    int c = lane + 32 * kk;
    h[(size_t)row * DM + c] = (v[kk] - mean) * rstd * gamma[c] + beta[c];
  }
}

__global__ void copy_cols_kernel(const float* __restrict__ src, int scols,
                                 float* __restrict__ dst, int dcols, int dofs, int n)
{
  int i = blockIdx.x * blockDim.x + threadIdx.x;
  if (i >= n) return;
  int r = i / scols, c = i % scols;
  dst[(size_t)r * dcols + dofs + c] = src[i];
}

__global__ void add_kernel(const float* __restrict__ a, const float* __restrict__ b,
                           float* __restrict__ o, int n)
{
  int i = blockIdx.x * blockDim.x + threadIdx.x;
  if (i < n) o[i] = a[i] + b[i];
}

// goal_sample[..., :2] from mean(Bx40x2, z=0), intention(Bx40x3), sim(Bx40)
__global__ void future_vectors_kernel(const float* __restrict__ mean,
                                      const float* __restrict__ intent,
                                      const float* __restrict__ sim,
                                      float* __restrict__ goal)
{
  int i = blockIdx.x * blockDim.x + threadIdx.x;
  if (i >= BB * 40) return;
  int b = i / 40, t = i % 40;
  if (t == 0) {
    goal[(size_t)b * 80 + 0] = mean[(size_t)b * 80 + 0];
    goal[(size_t)b * 80 + 1] = mean[(size_t)b * 80 + 1];
  } else {
    float ax = mean[(size_t)b * 80 + (t - 1) * 2 + 0];
    float ay = mean[(size_t)b * 80 + (t - 1) * 2 + 1];   // az = 0
    float cx = intent[(size_t)b * 120 + t * 3 + 0];
    float cy = intent[(size_t)b * 120 + t * 3 + 1];
    float cz = intent[(size_t)b * 120 + t * 3 + 2];
    float d  = sim[(size_t)b * 40 + t];
    float an  = sqrtf(ax * ax + ay * ay);
    float cn2 = cx * cx + cy * cy + cz * cz;
    float bn  = sqrtf(cn2 + d * d) / an;
    float bdx = ay * cz;
    float bdy = -ax * cz;
    float bdz = ax * cy - ay * cx;
    float bl  = sqrtf(bdx * bdx + bdy * bdy + bdz * bdz);
    goal[(size_t)b * 80 + t * 2 + 0] = bn * bdx / bl;
    goal[(size_t)b * 80 + t * 2 + 1] = bn * bdy / bl;
  }
}

// ---------------------------------------------------------------------------
extern "C" void kernel_launch(void* const* d_in, const int* in_sizes, int n_in,
                              void* d_out, int out_size, void* d_ws, size_t ws_size,
                              hipStream_t stream)
{
  (void)in_sizes; (void)n_in; (void)out_size; (void)ws_size;
  auto F = [&](int i) { return (const float*)d_in[i]; };

  const float* x      = F(0);   // B x 8 x 6  (== x[:, :, :6])
  const float* intent = F(1);   // B x 8 x 9
  const float* psim   = F(2);   // B x 8 x 3
  const float* mask   = F(3);   // B x B

  enum { EGO_INT = 4, EGO_MEAN = 8, EGO_SCALE = 12, EGO_SIM = 16, EGO_VAR = 20,
         INT_DEC = 24, INT_SOCIAL = 30, MEAN_DEC = 64, SCALE_DEC = 70,
         SCALE_ENC = 76, SIM_DEC = 82, SIM_SOCIAL = 88, SOCIAL = 122,
         VAR_DEC = 156 };
  // tlayer leaf offsets: 0 f1.W 1 f1.b 2 f2.W 3 f2.b 4 k.W 5 k.b 6 ln1_b
  //   7 ln1_g 8 ln2_b 9 ln2_g 10 o.W 11 o.b 12 q.W 13 q.b 14 v.W 15 v.b

  // ---- workspace carve-up (fp32 units) ----
  float* ws = (float*)d_ws;
  size_t off = 0;
  auto alloc = [&](size_t n) { float* p = ws + off; off += n; return p; };
  float* sH     = alloc((size_t)BB * DM);
  float* sQ     = alloc((size_t)BB * DM);
  float* sK     = alloc((size_t)BB * DM);
  float* sV     = alloc((size_t)BB * DM);
  float* sProj  = alloc((size_t)BB * DM);
  float* sFF    = alloc((size_t)BB * DM);   // also attention output buffer
  float* sH0    = alloc((size_t)BB * DM);
  float* outSoc = alloc((size_t)BB * DM);
  float* outInt = alloc((size_t)BB * DM);
  float* outSim = alloc((size_t)BB * DM);
  float* egoVar   = alloc((size_t)BB * DM);
  float* egoMean  = alloc((size_t)BB * DM);
  float* egoScale = alloc((size_t)BB * DM);
  float* egoInt   = alloc((size_t)BB * DM);
  float* egoSim   = alloc((size_t)BB * DM);
  float* stepHid  = alloc((size_t)BB * 512);
  float* cat      = alloc((size_t)BB * 544);
  float* dec1     = alloc((size_t)BB * 1024);  // aliases flash Apart (same size)
  float* dec2     = alloc((size_t)BB * 1024);  // aliases flash Mbuf/Lbuf
  float* sf4      = alloc((size_t)BB * 4);
  float* sf16     = alloc((size_t)BB * 16);
  float* sf32     = alloc((size_t)BB * 32);
  _Float16* qh = (_Float16*)alloc((size_t)BB * 128);   // BB*256 halfs
  _Float16* kh = (_Float16*)alloc((size_t)BB * 128);
  _Float16* vt = (_Float16*)alloc((size_t)BB * 128);
  _Float16* wtPool = (_Float16*)alloc(3600000);        // ~7.0M halfs of fp16 weights
  float* Apart = dec1;                      // KPART*BB*DM == BB*1024
  float* Mbuf  = dec2;                      // KPART*NHEADS*BB
  float* Lbuf  = dec2 + (size_t)KPART * NHEADS * BB;

  // ---- output carve-up ----
  float* out    = (float*)d_out;
  float* gvar   = out;                              // B x 1600
  float* gmean  = gvar  + (size_t)BB * 1600;        // B x 80
  float* gscale = gmean + (size_t)BB * 80;          // B x 1
  float* gint   = gscale + (size_t)BB * 1;          // B x 120
  float* gsim   = gint  + (size_t)BB * 120;         // B x 40
  float* ggoal  = gsim  + (size_t)BB * 40;          // B x 80

  // ---- pre-transpose+convert every GEMM weight to fp16 (MxK) ----
  _Float16* wt[162] = {};
  size_t wtoff = 0;
  auto cvt = [&](int idx, int K, int M) {
    _Float16* dst = wtPool + wtoff;
    wtoff = (wtoff + (size_t)K * M + 7) & ~(size_t)7;   // keep 16B alignment
    int n = K * M;
    wcvt_kernel<<<(n + 255) / 256, 256, 0, stream>>>(F(idx), dst, K, M);
    wt[idx] = dst;
  };
  {
    const int socBase[3] = { SOCIAL, INT_SOCIAL, SIM_SOCIAL };
    const int socIn[3]   = { 48, 72, 24 };
    for (int s = 0; s < 3; ++s) {
      cvt(socBase[s] + 0, socIn[s], DM);
      for (int l = 0; l < 2; ++l) {
        int tb = socBase[s] + 2 + l * 16;
        cvt(tb + 0, DM, DM); cvt(tb + 2, DM, DM);  cvt(tb + 4, DM, DM);
        cvt(tb + 10, DM, DM); cvt(tb + 12, DM, DM); cvt(tb + 14, DM, DM);
      }
    }
    cvt(EGO_VAR + 0, 6, 64);   cvt(EGO_VAR + 2, 512, DM);
    cvt(EGO_MEAN + 0, 6, 64);  cvt(EGO_MEAN + 2, 512, DM);
    cvt(EGO_SCALE + 0, 6, 64); cvt(EGO_SCALE + 2, 512, DM);
    cvt(EGO_INT + 0, 9, 64);   cvt(EGO_INT + 2, 512, DM);
    cvt(EGO_SIM + 0, 3, 64);   cvt(EGO_SIM + 2, 512, DM);
    cvt(INT_DEC + 0, 512, 256);  cvt(INT_DEC + 2, 256, 128);  cvt(INT_DEC + 4, 128, 120);
    cvt(SIM_DEC + 0, 512, 256);  cvt(SIM_DEC + 2, 256, 128);  cvt(SIM_DEC + 4, 128, 40);
    cvt(MEAN_DEC + 0, 512, 256); cvt(MEAN_DEC + 2, 256, 128); cvt(MEAN_DEC + 4, 128, 80);
    cvt(SCALE_DEC + 0, 512, 256); cvt(SCALE_DEC + 2, 256, 128); cvt(SCALE_DEC + 4, 128, 1);
    cvt(SCALE_ENC + 0, 1, 4);    cvt(SCALE_ENC + 2, 4, 16);   cvt(SCALE_ENC + 4, 16, 32);
    cvt(VAR_DEC + 0, 544, 1024); cvt(VAR_DEC + 2, 1024, 1024); cvt(VAR_DEC + 4, 1024, 1600);
  }

  // ---- launch helpers ----
  auto gemm = [&](const float* X, int widx, float* Y, int N, int K, int M, int act) {
    dim3 grid((M + 63) / 64, (N + 15) / 16);
    gemm_wmma_kernel<<<grid, dim3(32), 0, stream>>>(X, wt[widx], F(widx + 1), Y,
                                                    N, K, M, act);
  };
  auto copyc = [&](const float* src, int scols, float* dst, int dcols, int dofs) {
    int n = BB * scols;
    copy_cols_kernel<<<(n + 255) / 256, 256, 0, stream>>>(src, scols, dst, dcols, dofs, n);
  };

  auto tlayer = [&](int tb) {
    gemm(sH, tb + 12, sQ, BB, DM, DM, 0);   // q
    gemm(sH, tb + 4,  sK, BB, DM, DM, 0);   // k
    gemm(sH, tb + 14, sV, BB, DM, DM, 0);   // v
    pack_qkv_kernel<<<(BB * DM + 255) / 256, 256, 0, stream>>>(sQ, sK, sV, qh, kh, vt);
    flash_attn_kernel<<<dim3(BB / 16, NHEADS, KPART), dim3(32), 0, stream>>>(
        qh, kh, vt, mask, Apart, Mbuf, Lbuf);
    flash_combine_kernel<<<(BB * DM + 255) / 256, 256, 0, stream>>>(Apart, Mbuf, Lbuf, sFF);
    gemm(sFF, tb + 10, sProj, BB, DM, DM, 0);                         // o-proj
    ln_res_kernel<<<BB / 8, 256, 0, stream>>>(sH, sProj, F(tb + 7), F(tb + 6), BB);
    gemm(sH,  tb + 0, sFF,   BB, DM, DM, 1);                          // f1 + relu
    gemm(sFF, tb + 2, sProj, BB, DM, DM, 0);                          // f2
    ln_res_kernel<<<BB / 8, 256, 0, stream>>>(sH, sProj, F(tb + 9), F(tb + 8), BB);
  };

  auto social_fwd = [&](const float* xin, int indim, int base, float* outbuf) {
    gemm(xin, base + 0, sH0, BB, indim, DM, 0);                       // encoder
    copyc(sH0, DM, sH, DM, 0);
    tlayer(base + 2);        // l0
    tlayer(base + 18);       // l1
    add_kernel<<<(BB * DM + 255) / 256, 256, 0, stream>>>(sH0, sH, outbuf, BB * DM);
  };

  social_fwd(x,      48, SOCIAL,     outSoc);
  social_fwd(intent, 72, INT_SOCIAL, outInt);
  social_fwd(psim,   24, SIM_SOCIAL, outSim);

  auto stepwise = [&](const float* xin, int indim, int base, float* outbuf) {
    gemm(xin, base + 0, stepHid, BB * 8, indim, 64, 1);               // l1+relu
    gemm(stepHid, base + 2, outbuf, BB, 512, DM, 1);                  // l2+relu
  };
  stepwise(x,      6, EGO_VAR,   egoVar);
  stepwise(x,      6, EGO_MEAN,  egoMean);
  stepwise(x,      6, EGO_SCALE, egoScale);
  stepwise(intent, 9, EGO_INT,   egoInt);
  stepwise(psim,   3, EGO_SIM,   egoSim);

  auto mlp3 = [&](float* xin, int kin, int base, int m1, int m2, int m3, float* y) {
    gemm(xin,  base + 0, dec1, BB, kin, m1, 1);
    gemm(dec1, base + 2, dec2, BB, m1,  m2, 1);
    gemm(dec2, base + 4, y,    BB, m2,  m3, 0);
  };

  copyc(egoInt, DM, cat, 512, 0);  copyc(outInt, DM, cat, 512, DM);
  mlp3(cat, 512, INT_DEC, 256, 128, 120, gint);

  copyc(egoSim, DM, cat, 512, 0);  copyc(outSim, DM, cat, 512, DM);
  mlp3(cat, 512, SIM_DEC, 256, 128, 40, gsim);

  copyc(egoMean, DM, cat, 512, 0); copyc(outSoc, DM, cat, 512, DM);
  mlp3(cat, 512, MEAN_DEC, 256, 128, 80, gmean);

  copyc(egoScale, DM, cat, 512, 0); copyc(outSoc, DM, cat, 512, DM);
  mlp3(cat, 512, SCALE_DEC, 256, 128, 1, gscale);

  gemm(gscale, SCALE_ENC + 0, sf4,  BB, 1,  4,  1);
  gemm(sf4,    SCALE_ENC + 2, sf16, BB, 4,  16, 1);
  gemm(sf16,   SCALE_ENC + 4, sf32, BB, 16, 32, 0);

  copyc(egoVar, DM, cat, 544, 0); copyc(outSoc, DM, cat, 544, DM);
  copyc(sf32, 32, cat, 544, 512);
  mlp3(cat, 544, VAR_DEC, 1024, 1024, 1600, gvar);

  future_vectors_kernel<<<(BB * 40 + 255) / 256, 256, 0, stream>>>(gmean, gint, gsim, ggoal);
}
